// Seq2SeqModel_80204219285883
// MI455X (gfx1250) — matrix operand order repaired and checked by reading, not compiled, not run on hardware
//
#include <hip/hip_runtime.h>
#include <hip/hip_bf16.h>

// Problem constants
#define Hh   256
#define Ee   256
#define VOv  32000
#define Bb   64
#define Ss   128
#define Tt   64
#define G4H  1024   // 4*H
#define ENC_M (Ss*Bb)      // 8192
#define DEC_M ((Tt-1)*Bb)  // 4032

typedef __attribute__((ext_vector_type(8)))  _Float16 v8h;
typedef __attribute__((ext_vector_type(16))) _Float16 v16h;
typedef __attribute__((ext_vector_type(8)))  float    v8f;

// ---------------- WMMA fragment helpers (A/B use contiguous k = 16h + j, same order in both) ----------------

// A: 16x32 f16 tile from row-major A (lda halfs). Lane L: row = m0 + (L&15),
// halves j=0..15 -> K = klo+j, klo = k0 + 16*(L>=16).
__device__ __forceinline__ v16h load_frag_a(const _Float16* A, int lda, int m0, int k0, int lane) {
  int row = m0 + (lane & 15);
  int klo = k0 + ((lane >> 4) << 4);
  const _Float16* p = A + (size_t)row * lda + klo;
  v8h lo = *(const v8h*)p;
  v8h hi = *(const v8h*)(p + 8);
  v16h r;
#pragma unroll
  for (int j = 0; j < 8; ++j) { r[j] = lo[j]; r[j + 8] = hi[j]; }
  return r;
}

// B: 32x16 f16 tile where B[k][n] = W[n0+n][k0+k], W row-major (ldw halfs).
// Lane L: n = L&15, K = kb+j for j=0..15, kb = k0 + 16*(L>=16). Contiguous 16 halfs.
__device__ __forceinline__ v16h load_frag_b(const _Float16* W, int ldw, int n0, int k0, int lane) {
  int row = n0 + (lane & 15);
  int kb = k0 + ((lane >> 4) << 4);
  const _Float16* p = W + (size_t)row * ldw + kb;
  v8h lo = *(const v8h*)p;
  v8h hi = *(const v8h*)(p + 8);
  v16h r;
#pragma unroll
  for (int j = 0; j < 8; ++j) { r[j] = lo[j]; r[j + 8] = hi[j]; }
  return r;
}

__device__ __forceinline__ v8f wmma16(v16h a, v16h b, v8f c) {
  return __builtin_amdgcn_wmma_f32_16x16x32_f16(false, a, false, b, (short)0, c, false, false);
}

// C/D element (r, lane): M = r + 8*(lane>=16), N = lane&15
#define C_ROW(m0, r, lane) ((m0) + (r) + (((lane) >> 4) << 3))
#define C_COL(n0, lane)    ((n0) + ((lane) & 15))

// ---------------- small utility kernels ----------------

__global__ void k_f32_to_f16(const float* __restrict__ s, _Float16* __restrict__ d, int n) {
  int i = blockIdx.x * blockDim.x + threadIdx.x;
  if (i < n) d[i] = (_Float16)s[i];
}

// enc_x row m = s*64+b ; token = src[b*S+s]
__global__ void k_embed_enc(const int* __restrict__ src, const float* __restrict__ emb,
                            _Float16* __restrict__ X) {
  int idx = blockIdx.x * blockDim.x + threadIdx.x;
  if (idx >= ENC_M * Ee) return;
  int e = idx & (Ee - 1);
  int m = idx >> 8;
  int b = m & (Bb - 1);
  int s = m >> 6;
  int tok = src[b * Ss + s];
  X[idx] = (_Float16)emb[(size_t)tok * Ee + e];
}

// dec_x row m = t*64+b ; token = tgt[b*T+t], t in [0,63)
__global__ void k_embed_dec(const int* __restrict__ tgt, const float* __restrict__ emb,
                            _Float16* __restrict__ X) {
  int idx = blockIdx.x * blockDim.x + threadIdx.x;
  if (idx >= DEC_M * Ee) return;
  int e = idx & (Ee - 1);
  int m = idx >> 8;
  int b = m & (Bb - 1);
  int t = m >> 6;
  int tok = tgt[b * Tt + t];
  X[idx] = (_Float16)emb[(size_t)tok * Ee + e];
}

__global__ void k_zero_t0(float* __restrict__ out) {
  int i = blockIdx.x * blockDim.x + threadIdx.x;
  if (i >= Bb * VOv) return;
  int b = i / VOv;
  int v = i - b * VOv;
  out[(size_t)b * Tt * VOv + v] = 0.0f;
}

// ---------------- batched input projection: C = A @ W^T + bias ----------------
// A: M x K f16 row-major, W: N x K f16 row-major, C: M x N f32.
// One wave = 32x64 tile (2 m-tiles x 4 n-tiles, A fragments reused across B).
__global__ void k_gemm_xproj(const _Float16* __restrict__ A, const _Float16* __restrict__ W,
                             const float* __restrict__ bias, float* __restrict__ C,
                             int M, int N, int K) {
  int gw = (blockIdx.x * blockDim.x + threadIdx.x) >> 5;
  int lane = threadIdx.x & 31;
  int nGroups = N >> 6;
  int m0 = (gw / nGroups) << 5;
  int n0base = (gw % nGroups) << 6;
  if (m0 >= M) return;

  v8f acc[2][4] = {};
  for (int k0 = 0; k0 < K; k0 += 32) {
    if (k0 + 32 < K)
      __builtin_prefetch(A + (size_t)(m0 + (lane & 15)) * K + k0 + 32, 0, 1);
    v16h a0 = load_frag_a(A, K, m0, k0, lane);
    v16h a1 = load_frag_a(A, K, m0 + 16, k0, lane);
#pragma unroll
    for (int nt = 0; nt < 4; ++nt) {
      v16h b = load_frag_b(W, K, n0base + nt * 16, k0, lane);
      acc[0][nt] = wmma16(a0, b, acc[0][nt]);
      acc[1][nt] = wmma16(a1, b, acc[1][nt]);
    }
  }
#pragma unroll
  for (int mt = 0; mt < 2; ++mt) {
#pragma unroll
    for (int nt = 0; nt < 4; ++nt) {
      int n = C_COL(n0base + nt * 16, lane);
      float bv = bias[n];
#pragma unroll
      for (int r = 0; r < 8; ++r) {
        int m = C_ROW(m0 + mt * 16, r, lane);
        C[(size_t)m * N + n] = acc[mt][nt][r] + bv;
      }
    }
  }
}

// ---------------- persistent single-WGP LSTM recurrence ----------------
// One block, 1024 threads = 32 wave32 waves. LDS: gates 64x1024 f32 (256KB) + h 64x256 f16 (32KB).
// (320KB LDS per WGP is CDNA5-only; this cannot fit on CDNA4.)
// gates LDS holds only h @ Whh^T; the precomputed Xih[t] (+bias) is added in the
// elementwise phase with fully-coalesced contiguous loads, keeping the serial
// per-step critical path short.
__global__ void __launch_bounds__(1024) k_lstm_recurrence(
    const float* __restrict__ XihE, const _Float16* __restrict__ WhhE,
    const float* __restrict__ XihD, const _Float16* __restrict__ WhhD,
    _Float16* __restrict__ hs) {
  extern __shared__ char smem[];
  float*    gates = (float*)smem;                              // 64*1024 f32
  _Float16* hbuf  = (_Float16*)(smem + (size_t)Bb * G4H * 4);  // 64*256 f16

  const int tid  = threadIdx.x;
  const int lane = tid & 31;
  const int wv   = tid >> 5;
  // GEMM tile assignment: wave -> (m-tile, 128-col strip): 4 m-tiles x 8 strips = 32 waves
  const int m0     = (wv >> 3) << 4;
  const int n0base = (wv & 7) << 7;
  // elementwise ownership: thread -> (b, 16 consecutive h)
  const int eb  = tid >> 4;
  const int hh0 = (tid & 15) << 4;

  float creg[16];
#pragma unroll
  for (int j = 0; j < 16; ++j) creg[j] = 0.0f;
  for (int i = tid; i < Bb * Hh; i += 1024) hbuf[i] = (_Float16)0.0f;
  __syncthreads();

  for (int phase = 0; phase < 2; ++phase) {
    const float*    Xih = phase ? XihD : XihE;
    const _Float16* Whh = phase ? WhhD : WhhE;
    const int nsteps = phase ? (Tt - 1) : Ss;

    for (int t = 0; t < nsteps; ++t) {
      // gates = h @ Whh^T   (M=64, N=1024, K=256)
      v8f acc[8] = {};
      for (int k0 = 0; k0 < Hh; k0 += 32) {
        v16h a = load_frag_a(hbuf, Hh, m0, k0, lane);
#pragma unroll
        for (int nt = 0; nt < 8; ++nt) {
          v16h b = load_frag_b(Whh, Hh, n0base + nt * 16, k0, lane);
          acc[nt] = wmma16(a, b, acc[nt]);
        }
      }
#pragma unroll
      for (int nt = 0; nt < 8; ++nt) {
        int n = C_COL(n0base + nt * 16, lane);
#pragma unroll
        for (int r = 0; r < 8; ++r) {
          int m = C_ROW(m0, r, lane);
          gates[m * G4H + n] = acc[nt][r];
        }
      }
      __syncthreads();

      // fused LSTM cell elementwise: gates += Xih[t] (coalesced), then nonlinearity
      const float* xrow = Xih + (size_t)(t * Bb + eb) * G4H;
#pragma unroll
      for (int j = 0; j < 16; ++j) {
        int hh = hh0 + j;
        float ig = gates[eb * G4H + hh]          + xrow[hh];
        float fg = gates[eb * G4H + Hh + hh]     + xrow[Hh + hh];
        float gg = gates[eb * G4H + 2 * Hh + hh] + xrow[2 * Hh + hh];
        float og = gates[eb * G4H + 3 * Hh + hh] + xrow[3 * Hh + hh];
        float si = 1.0f / (1.0f + __expf(-ig));
        float sf = 1.0f / (1.0f + __expf(-fg));
        float so = 1.0f / (1.0f + __expf(-og));
        float c  = sf * creg[j] + si * tanhf(gg);
        creg[j]  = c;
        float h  = so * tanhf(c);
        hbuf[eb * Hh + hh] = (_Float16)h;
        if (phase) hs[(size_t)(t * Bb + eb) * Hh + hh] = (_Float16)h;
      }
      __syncthreads();
    }
  }
}

// ---------------- final projection: out[b][t+1][v] = hs[t*64+b] . fc_w[v] + fc_b[v] ----------------
// One wave = 32x64 tile: halves redundant fc_w (B) traffic vs 16x64.
__global__ void k_final_gemm(const _Float16* __restrict__ HS, const _Float16* __restrict__ W,
                             const float* __restrict__ bias, float* __restrict__ out) {
  int gw = (blockIdx.x * blockDim.x + threadIdx.x) >> 5;
  int lane = threadIdx.x & 31;
  const int nGroups = VOv >> 6;  // 500
  int m0 = (gw / nGroups) << 5;
  int n0base = (gw % nGroups) << 6;
  if (m0 >= DEC_M) return;

  v8f acc[2][4] = {};
  for (int k0 = 0; k0 < Hh; k0 += 32) {
    if (k0 + 32 < Hh)
      __builtin_prefetch(W + (size_t)(n0base + (lane & 15)) * Hh + k0 + 32, 0, 1);
    v16h a0 = load_frag_a(HS, Hh, m0, k0, lane);
    v16h a1 = load_frag_a(HS, Hh, m0 + 16, k0, lane);
#pragma unroll
    for (int nt = 0; nt < 4; ++nt) {
      v16h b = load_frag_b(W, Hh, n0base + nt * 16, k0, lane);
      acc[0][nt] = wmma16(a0, b, acc[0][nt]);
      acc[1][nt] = wmma16(a1, b, acc[1][nt]);
    }
  }
#pragma unroll
  for (int mt = 0; mt < 2; ++mt) {
#pragma unroll
    for (int nt = 0; nt < 4; ++nt) {
      int n = C_COL(n0base + nt * 16, lane);
      float bv = bias[n];
#pragma unroll
      for (int r = 0; r < 8; ++r) {
        int m  = C_ROW(m0 + mt * 16, r, lane);
        int b  = m & (Bb - 1);
        int t  = m >> 6;
        out[((size_t)b * Tt + (t + 1)) * VOv + n] = acc[mt][nt][r] + bv;
      }
    }
  }
}

// ---------------- launch ----------------

extern "C" void kernel_launch(void* const* d_in, const int* in_sizes, int n_in,
                              void* d_out, int out_size, void* d_ws, size_t ws_size,
                              hipStream_t stream) {
  const int*   src     = (const int*)d_in[0];
  const int*   tgt     = (const int*)d_in[1];
  const float* enc_emb = (const float*)d_in[2];
  const float* Wih_e   = (const float*)d_in[3];
  const float* Whh_e   = (const float*)d_in[4];
  const float* b_e     = (const float*)d_in[5];
  const float* dec_emb = (const float*)d_in[6];
  const float* Wih_d   = (const float*)d_in[7];
  const float* Whh_d   = (const float*)d_in[8];
  const float* b_d     = (const float*)d_in[9];
  const float* fc_w    = (const float*)d_in[10];
  const float* fc_b    = (const float*)d_in[11];
  float* out = (float*)d_out;

  char* ws = (char*)d_ws;
  _Float16* WihE_h = (_Float16*)(ws + 0);          //  512 KB
  _Float16* WhhE_h = (_Float16*)(ws + 524288);     //  512 KB
  _Float16* WihD_h = (_Float16*)(ws + 1048576);    //  512 KB
  _Float16* WhhD_h = (_Float16*)(ws + 1572864);    //  512 KB
  _Float16* fcw_h  = (_Float16*)(ws + 2097152);    // 16.0 MB  (32000*256*2)
  _Float16* enc_x  = (_Float16*)(ws + 18481152);   //  4.0 MB  (8192*256*2)
  _Float16* dec_x  = (_Float16*)(ws + 22675456);   //  2.0 MB  (4032*256*2)
  float*    Xih_e  = (float*)   (ws + 24739840);   // 32.0 MB  (8192*1024*4)
  float*    Xih_d  = (float*)   (ws + 58294272);   // 15.8 MB  (4032*1024*4)
  _Float16* hs     = (_Float16*)(ws + 74809344);   //  2.0 MB  (4032*256*2)

  // 1) weights fp32 -> f16
  k_f32_to_f16<<<(G4H * Ee + 255) / 256, 256, 0, stream>>>(Wih_e, WihE_h, G4H * Ee);
  k_f32_to_f16<<<(G4H * Hh + 255) / 256, 256, 0, stream>>>(Whh_e, WhhE_h, G4H * Hh);
  k_f32_to_f16<<<(G4H * Ee + 255) / 256, 256, 0, stream>>>(Wih_d, WihD_h, G4H * Ee);
  k_f32_to_f16<<<(G4H * Hh + 255) / 256, 256, 0, stream>>>(Whh_d, WhhD_h, G4H * Hh);
  k_f32_to_f16<<<(VOv * Hh + 255) / 256, 256, 0, stream>>>(fc_w, fcw_h, VOv * Hh);

  // 2) embedding gathers (time-major rows m = t*B + b)
  k_embed_enc<<<(ENC_M * Ee) / 256, 256, 0, stream>>>(src, enc_emb, enc_x);
  k_embed_dec<<<(DEC_M * Ee) / 256, 256, 0, stream>>>(tgt, dec_emb, dec_x);

  // 3) batched input projections (bias folded in): Xih = X @ Wih^T + b
  {
    int waves = (ENC_M / 32) * (G4H / 64);          // 4096 waves
    k_gemm_xproj<<<waves * 32 / 256, 256, 0, stream>>>(enc_x, WihE_h, b_e, Xih_e, ENC_M, G4H, Ee);
  }
  {
    int waves = (DEC_M / 32) * (G4H / 64);          // 2016 waves
    k_gemm_xproj<<<waves * 32 / 256, 256, 0, stream>>>(dec_x, WihD_h, b_d, Xih_d, DEC_M, G4H, Ee);
  }

  // 4) persistent recurrence: 1 block, 32 waves, 288KB dynamic LDS (gates + h)
  size_t lds_bytes = (size_t)Bb * G4H * 4 + (size_t)Bb * Hh * 2;  // 294912
  k_lstm_recurrence<<<1, 1024, lds_bytes, stream>>>(Xih_e, WhhE_h, Xih_d, WhhD_h, hs);

  // 5) output projection + scatter: 126 m-groups x 500 col-groups = 63000 waves
  {
    int waves = (DEC_M / 32) * (VOv / 64);
    k_final_gemm<<<waves * 32 / 256, 256, 0, stream>>>(hs, fcw_h, fc_b, out);
  }

  // 6) zero t=0 slice
  k_zero_t0<<<(Bb * VOv + 255) / 256, 256, 0, stream>>>(out);
}
